// SwinBlock_7249904795782
// MI455X (gfx1250) — compile-verified, matrix operand order checked by reading, not run on hardware
//
#include <hip/hip_runtime.h>
#include <hip/hip_bf16.h>
#include <math.h>

typedef __bf16 bf16;
typedef __attribute__((ext_vector_type(16))) __bf16 v16bf;
typedef __attribute__((ext_vector_type(8)))  float  v8f;
typedef __attribute__((ext_vector_type(4)))  unsigned int u32x4;
typedef __attribute__((ext_vector_type(8)))  int i32x8;
typedef __attribute__((ext_vector_type(4)))  int i32x4;

#if defined(__has_include)
#if __has_include(<hip/amd_detail/amd_gfx1250_TDM.h>)
#define TDM_6ARG 1
#endif
#endif

#define WMMA_BF16(a, b, c) \
    __builtin_amdgcn_wmma_f32_16x16x32_bf16(false, (a), false, (b), (short)0, (c), false, false)

// ---------------------------------------------------------------------------
// Fragment loader: lane L holds row (L%16), K = (L/16)*8 + {0..7} and +16.
// Two 16B loads (global_load_b128 / ds_load_b128 depending on address space).
// ---------------------------------------------------------------------------
__device__ __forceinline__ v16bf frag_ld(const bf16* p) {
    v16bf f;
    uint4* u = reinterpret_cast<uint4*>(&f);
    u[0] = *reinterpret_cast<const uint4*>(p);
    u[1] = *reinterpret_cast<const uint4*>(p + 16);
    return f;
}

// ---------------------------------------------------------------------------
// fp32 -> bf16 conversion (weights)
// ---------------------------------------------------------------------------
__global__ void __launch_bounds__(256)
f2bf_kernel(const float* __restrict__ in, bf16* __restrict__ out, int n) {
    int i = blockIdx.x * 256 + threadIdx.x;
    if (i < n) out[i] = (bf16)in[i];
}

// ---------------------------------------------------------------------------
// LayerNorm over C=384: one wave per token, 12 elems per lane, shfl reduce.
// ---------------------------------------------------------------------------
__global__ void __launch_bounds__(256)
ln_kernel(const float* __restrict__ x, const float* __restrict__ w,
          const float* __restrict__ b, bf16* __restrict__ out) {
    const int lane = threadIdx.x & 31;
    const int wave = threadIdx.x >> 5;
    const size_t tok = (size_t)blockIdx.x * 8 + wave;
    const float* xp = x + tok * 384;
    float v[12];
    float s = 0.f;
#pragma unroll
    for (int j = 0; j < 12; ++j) { v[j] = xp[lane + 32 * j]; s += v[j]; }
#pragma unroll
    for (int o = 16; o > 0; o >>= 1) s += __shfl_xor(s, o, 32);
    const float mean = s * (1.0f / 384.0f);
    float vs = 0.f;
#pragma unroll
    for (int j = 0; j < 12; ++j) { float d = v[j] - mean; vs += d * d; }
#pragma unroll
    for (int o = 16; o > 0; o >>= 1) vs += __shfl_xor(vs, o, 32);
    const float rstd = rsqrtf(vs * (1.0f / 384.0f) + 1e-5f);
    bf16* op = out + tok * 384;
#pragma unroll
    for (int j = 0; j < 12; ++j) {
        int c = lane + 32 * j;
        op[c] = (bf16)((v[j] - mean) * rstd * w[c] + b[c]);
    }
}

// ---------------------------------------------------------------------------
// Generic WMMA GEMM: out[M,N] = A[M,K](bf16) @ W[N,K]^T(bf16) + bias, epilogue:
//   mode 0: bf16 out, cols < scale_cols multiplied by scale (QKV, q*SCALE)
//   mode 1: f32 out                                          (proj -> y)
//   mode 2: bf16 out, exact GELU                             (fc1)
//   mode 3: f32 out + residual                               (fc2 + y)
// Each wave: 16x64 tile; all 8 waves of a block share the 64-row W panel, so
// the 64x32 W tile is staged in LDS ONCE per block via the Tensor Data Mover
// (double-buffered; TDM applies the 80B padded row stride for conflict-free
// ds_load_b128 fragment reads). Wave 0 drives the DMA + s_wait_tensorcnt.
// ---------------------------------------------------------------------------
__device__ __forceinline__ void tdm_load_w_tile(const bf16* gsrc, unsigned lds_addr, int K) {
    const unsigned long long ga = (unsigned long long)(uintptr_t)gsrc;
    // D# group 0: count=1 | lds_addr | global_addr[56:0] | type=2
    u32x4 g0 = { 1u, lds_addr, (unsigned)ga, (unsigned)(ga >> 32) | (2u << 30) };
    // D# group 1: data_size=2B, pad_enable, pad_interval=16 DW (per 32-elem row),
    // pad_amount=4 DW (-> 80B row stride); tensor dims large (never OOB);
    // tile_dim0=32 (K slice), tile_dim1=64 (rows); stride0 = K elements.
    const unsigned DIM = 1u << 18;
    i32x8 g1 = { (int)((1u << 16) | (1u << 20) | (3u << 22) | (3u << 25)),
                 (int)((DIM & 0xffffu) << 16),
                 (int)((DIM >> 16) | ((DIM & 0xffffu) << 16)),
                 (int)((DIM >> 16) | (32u << 16)),
                 (int)64,
                 (int)K,
                 0, 0 };
    i32x4 gz4 = { 0, 0, 0, 0 };
#ifdef TDM_6ARG
    i32x8 gz8 = { 0, 0, 0, 0, 0, 0, 0, 0 };
    __builtin_amdgcn_tensor_load_to_lds(g0, g1, gz4, gz4, gz8, 0);
#else
    __builtin_amdgcn_tensor_load_to_lds(g0, g1, gz4, gz4, 0);
#endif
}

__global__ void __launch_bounds__(256)
gemm_wmma_kernel(const bf16* __restrict__ A, const bf16* __restrict__ W,
                 const float* __restrict__ bias, void* __restrict__ outp,
                 const float* __restrict__ resid,
                 int K, int N, int mode, int scale_cols, float scale_val) {
    __shared__ __attribute__((aligned(128))) bf16 Ws[2][64 * 40];  // 80B rows
    const int lane = threadIdx.x & 31;
    const int wave = threadIdx.x >> 5;
    const int m0 = (blockIdx.y * 8 + wave) * 16;
    const int n0 = blockIdx.x * 64;
    const int lrow = lane & 15;
    const int koff = (lane >> 4) << 3;
    const int ntiles = K >> 5;

    const bf16* pa = A + (size_t)(m0 + lrow) * K + koff;
    v8f acc[4] = {};

    if (wave == 0)
        tdm_load_w_tile(W + (size_t)n0 * K, (unsigned)(uintptr_t)&Ws[0][0], K);

    for (int i = 0; i < ntiles; ++i) {
        if (wave == 0) {
            if (i + 1 < ntiles) {
                tdm_load_w_tile(W + (size_t)n0 * K + ((i + 1) << 5),
                                (unsigned)(uintptr_t)&Ws[(i + 1) & 1][0], K);
                __builtin_amdgcn_s_wait_tensorcnt((short)1);  // current buf done
            } else {
                __builtin_amdgcn_s_wait_tensorcnt((short)0);
            }
        }
        __syncthreads();  // publish Ws[i&1] to all waves
        const bf16* ws = &Ws[i & 1][0];
        __builtin_prefetch(pa + (i << 5) + 128, 0, 1);
        v16bf a  = frag_ld(pa + (i << 5));
        v16bf b0 = frag_ld(ws + ( 0 + lrow) * 40 + koff);
        v16bf b1 = frag_ld(ws + (16 + lrow) * 40 + koff);
        v16bf b2 = frag_ld(ws + (32 + lrow) * 40 + koff);
        v16bf b3 = frag_ld(ws + (48 + lrow) * 40 + koff);
        acc[0] = WMMA_BF16(a, b0, acc[0]);
        acc[1] = WMMA_BF16(a, b1, acc[1]);
        acc[2] = WMMA_BF16(a, b2, acc[2]);
        acc[3] = WMMA_BF16(a, b3, acc[3]);
        __syncthreads();  // all reads of Ws[i&1] done before it is refilled
    }

    const int rbase = m0 + ((lane >> 4) << 3);
#pragma unroll
    for (int t = 0; t < 4; ++t) {
        const int col = n0 + t * 16 + lrow;
        const float bv = bias[col];
        const float sc = (mode == 0 && col < scale_cols) ? scale_val : 1.0f;
#pragma unroll
        for (int r = 0; r < 8; ++r) {
            const size_t idx = (size_t)(rbase + r) * N + col;
            float v = acc[t][r] + bv;
            if (mode == 0) {
                ((bf16*)outp)[idx] = (bf16)(v * sc);
            } else if (mode == 1) {
                ((float*)outp)[idx] = v;
            } else if (mode == 2) {
                ((bf16*)outp)[idx] = (bf16)(0.5f * v * (1.0f + erff(v * 0.70710678f)));
            } else {
                ((float*)outp)[idx] = v + resid[idx];
            }
        }
    }
}

// ---------------------------------------------------------------------------
// Windowed attention: one block per (window, head), 4 waves (128 threads).
// S = (q*SCALE) k^T + relbias (one WMMA per 16x16 tile, K=32 exact),
// softmax over 49 cols, O = P v, scatter with window-reverse indexing.
// ---------------------------------------------------------------------------
__global__ void __launch_bounds__(128)
attn_kernel(const bf16* __restrict__ qkv, const float* __restrict__ btab,
            bf16* __restrict__ out) {
    constexpr int SQ = 40;  // Qs/Ks row stride (bf16): conflict-free b128 reads
    constexpr int SV = 72;  // Vt / Ps row stride (bf16), 16B-aligned rows
    constexpr int SS = 65;  // S row stride (f32), conflict-free row scans
    __shared__ __attribute__((aligned(16))) bf16 Qs[64 * SQ];
    __shared__ __attribute__((aligned(16))) bf16 Ks[64 * SQ];
    __shared__ __attribute__((aligned(16))) bf16 Vt[32 * SV];   // transposed V
    __shared__ __attribute__((aligned(16))) bf16 Ps[64 * SV];
    __shared__ __attribute__((aligned(16))) float Sf[64 * SS];

    const int win = blockIdx.x;
    const int head = blockIdx.y;
    const int bimg = win >> 6;       // window / 64
    const int wy = (win >> 3) & 7;
    const int wx = win & 7;
    const int tid = threadIdx.x;
    const int lane = tid & 31;
    const int wave = tid >> 5;
    const int cbase = head * 32;

    auto gtok = [&](int n) -> size_t {   // window token -> global token row
        int iy = n / 7, ix = n - iy * 7;
        return (size_t)bimg * 3136 + (size_t)(wy * 7 + iy) * 56 + (wx * 7 + ix);
    };

    // ---- stage Q, K (rows >= 49 zero-padded) ----
    for (int j = tid; j < 256; j += 128) {
        const int row = j >> 2, seg = j & 3;
        uint4 zq = {0, 0, 0, 0}, zk = {0, 0, 0, 0};
        if (row < 49) {
            const size_t g = gtok(row);
            zq = *reinterpret_cast<const uint4*>(qkv + g * 1152 + 0   + cbase + seg * 8);
            zk = *reinterpret_cast<const uint4*>(qkv + g * 1152 + 384 + cbase + seg * 8);
        }
        *reinterpret_cast<uint4*>(&Qs[row * SQ + seg * 8]) = zq;
        *reinterpret_cast<uint4*>(&Ks[row * SQ + seg * 8]) = zk;
    }
    // ---- stage V transposed: Vt[d][m] (m >= 49 zeroed for NaN-safe WMMA) ----
    {
        const int row = tid >> 1, half = tid & 1;
        bf16 tmp[16];
        if (row < 49) {
            const size_t g = gtok(row);
            const bf16* v = qkv + g * 1152 + 768 + cbase + half * 16;
            *reinterpret_cast<uint4*>(&tmp[0]) = *reinterpret_cast<const uint4*>(v);
            *reinterpret_cast<uint4*>(&tmp[8]) = *reinterpret_cast<const uint4*>(v + 8);
        } else {
#pragma unroll
            for (int i = 0; i < 16; ++i) tmp[i] = (bf16)0.f;
        }
#pragma unroll
        for (int i = 0; i < 16; ++i) Vt[(half * 16 + i) * SV + row] = tmp[i];
    }
    __syncthreads();

    // ---- S = Q K^T + bias (wave handles 16-row block, 4 col tiles) ----
    const int lrow = lane & 15;
    const int koff = (lane >> 4) << 3;
    const v16bf aq = frag_ld(&Qs[(wave * 16 + lrow) * SQ + koff]);
#pragma unroll
    for (int nt = 0; nt < 4; ++nt) {
        v16bf bk = frag_ld(&Ks[(nt * 16 + lrow) * SQ + koff]);
        v8f s = {};
        s = WMMA_BF16(aq, bk, s);
        const int colc = nt * 16 + lrow;
        const int rb = wave * 16 + ((lane >> 4) << 3);
#pragma unroll
        for (int r = 0; r < 8; ++r) {
            const int rowc = rb + r;
            float val = -1e30f;
            if (rowc < 49 && colc < 49) {
                int iy = rowc / 7, ix = rowc - iy * 7;
                int my = colc / 7, mx = colc - my * 7;
                int ridx = (iy - my + 6) * 13 + (ix - mx + 6);
                val = s[r] + btab[ridx * 12 + head];
            }
            Sf[rowc * SS + colc] = val;
        }
    }
    __syncthreads();

    // ---- softmax per row; P in bf16 (K-padding columns zeroed) ----
    if (tid < 64) {
        const int rowc = tid;
        if (rowc < 49) {
            float mx = -1e30f;
            for (int m = 0; m < 49; ++m) mx = fmaxf(mx, Sf[rowc * SS + m]);
            float sum = 0.f;
            for (int m = 0; m < 49; ++m) sum += __expf(Sf[rowc * SS + m] - mx);
            const float inv = 1.0f / sum;
            for (int m = 0; m < 49; ++m)
                Ps[rowc * SV + m] = (bf16)(__expf(Sf[rowc * SS + m] - mx) * inv);
            for (int m = 49; m < 64; ++m) Ps[rowc * SV + m] = (bf16)0.f;
        } else {
            for (int m = 0; m < 64; ++m) Ps[rowc * SV + m] = (bf16)0.f;
        }
    }
    __syncthreads();

    // ---- O = P V  (K = 64 padded -> 2 WMMA steps, 2 d-tiles) ----
    v8f o0 = {}, o1 = {};
#pragma unroll
    for (int kb = 0; kb < 64; kb += 32) {
        v16bf ap = frag_ld(&Ps[(wave * 16 + lrow) * SV + kb + koff]);
        v16bf b0 = frag_ld(&Vt[( 0 + lrow) * SV + kb + koff]);
        v16bf b1 = frag_ld(&Vt[(16 + lrow) * SV + kb + koff]);
        o0 = WMMA_BF16(ap, b0, o0);
        o1 = WMMA_BF16(ap, b1, o1);
    }
    const int rb = wave * 16 + ((lane >> 4) << 3);
#pragma unroll
    for (int t = 0; t < 2; ++t) {
        const int d = t * 16 + lrow;
        const v8f& o = t ? o1 : o0;
#pragma unroll
        for (int r = 0; r < 8; ++r) {
            const int n = rb + r;
            if (n < 49) out[gtok(n) * 384 + cbase + d] = (bf16)o[r];
        }
    }
}

// ---------------------------------------------------------------------------
// Host-side orchestration.
// ---------------------------------------------------------------------------
extern "C" void kernel_launch(void* const* d_in, const int* in_sizes, int n_in,
                              void* d_out, int out_size, void* d_ws, size_t ws_size,
                              hipStream_t stream) {
    (void)in_sizes; (void)n_in; (void)out_size; (void)ws_size;
    const int M = 32 * 56 * 56;                 // 100352 tokens
    const float SCALE = 0.17677669529663687f;   // 32^-0.5

    const float* x      = (const float*)d_in[0];
    const float* n1w    = (const float*)d_in[1];
    const float* n1b    = (const float*)d_in[2];
    const float* qkv_wf = (const float*)d_in[3];
    const float* qkv_b  = (const float*)d_in[4];
    const float* prj_wf = (const float*)d_in[5];
    const float* prj_b  = (const float*)d_in[6];
    const float* rtab   = (const float*)d_in[7];
    const float* n2w    = (const float*)d_in[8];
    const float* n2b    = (const float*)d_in[9];
    const float* fc1_wf = (const float*)d_in[10];
    const float* fc1_b  = (const float*)d_in[11];
    const float* fc2_wf = (const float*)d_in[12];
    const float* fc2_b  = (const float*)d_in[13];

    // workspace layout (region A reused for fc1 activations after attention)
    char* ws = (char*)d_ws;
    bf16* h1   = (bf16*)ws;                                  // [M,384]
    bf16* qkvb = (bf16*)(ws + (size_t)M * 384 * 2);          // [M,1152]
    bf16* a1   = (bf16*)ws;                                  // [M,1536] alias
    const size_t offB = (size_t)M * 1536 * 2;
    bf16* attnout = (bf16*)(ws + offB);                      // [M,384]
    bf16* h2      = attnout;                                 // alias after proj
    const size_t offC = offB + (size_t)M * 384 * 2;
    float* y = (float*)(ws + offC);                          // [M,384] f32
    const size_t offD = offC + (size_t)M * 384 * 4;
    bf16* wq = (bf16*)(ws + offD);
    bf16* wp = wq + 1152 * 384;
    bf16* w1 = wp + 384 * 384;
    bf16* w2 = w1 + 1536 * 384;

    f2bf_kernel<<<(1152 * 384 + 255) / 256, 256, 0, stream>>>(qkv_wf, wq, 1152 * 384);
    f2bf_kernel<<<( 384 * 384 + 255) / 256, 256, 0, stream>>>(prj_wf, wp,  384 * 384);
    f2bf_kernel<<<(1536 * 384 + 255) / 256, 256, 0, stream>>>(fc1_wf, w1, 1536 * 384);
    f2bf_kernel<<<( 384 *1536 + 255) / 256, 256, 0, stream>>>(fc2_wf, w2,  384 * 1536);

    ln_kernel<<<M / 8, 256, 0, stream>>>(x, n1w, n1b, h1);

    gemm_wmma_kernel<<<dim3(1152 / 64, M / 128), 256, 0, stream>>>(
        h1, wq, qkv_b, qkvb, nullptr, 384, 1152, /*mode=*/0, /*scale_cols=*/384, SCALE);

    attn_kernel<<<dim3(2048, 12), 128, 0, stream>>>(qkvb, rtab, attnout);

    gemm_wmma_kernel<<<dim3(384 / 64, M / 128), 256, 0, stream>>>(
        attnout, wp, prj_b, y, nullptr, 384, 384, /*mode=*/1, 0, 1.0f);

    ln_kernel<<<M / 8, 256, 0, stream>>>(y, n2w, n2b, h2);

    gemm_wmma_kernel<<<dim3(1536 / 64, M / 128), 256, 0, stream>>>(
        h2, w1, fc1_b, a1, nullptr, 384, 1536, /*mode=*/2, 0, 1.0f);

    gemm_wmma_kernel<<<dim3(384 / 64, M / 128), 256, 0, stream>>>(
        a1, w2, fc2_b, (float*)d_out, y, 1536, 384, /*mode=*/3, 0, 1.0f);
}
